// MyConv1d_ell_2_64037962383413
// MI455X (gfx1250) — matrix operand order repaired and checked by reading, not compiled, not run on hardware
//
#include <hip/hip_runtime.h>

typedef _Float16 v16h __attribute__((ext_vector_type(16)));
typedef _Float16 v4h  __attribute__((ext_vector_type(4)));
typedef float    v8f  __attribute__((ext_vector_type(8)));

#define C_IN   1024
#define O_OUT  1024
#define BATCH  2048
#define OUTW   54
#define FRAGS2 (64*64)   /* K=2048 -> 64 kfrags, O=1024 -> 64 nfrags */
#define FRAGS3 (96*64)   /* K=3072 -> 96 kfrags */

// -------- position / weight tables: [which][slot][tap][entry] --------
// which=0: GEMM-A (lengths 4,5,6 via kk / filter_2), 15 slots, <=2 entries
// which=1: GEMM-B (lengths 6..9 via pairs / filter_3), 10 slots, <=3 entries
// Unused entries have weight 0 and position 0 -> gather is branch-free.
__constant__ int g_pos[2][16][3][3] = {
  { // GEMM-A
    {{0,0,0},{4,0,0},{0,0,0}},  {{2,0,0},{6,0,0},{0,0,0}},
    {{4,0,0},{8,0,0},{0,0,0}},  {{6,0,0},{10,0,0},{0,0,0}},
    {{8,0,0},{12,0,0},{0,0,0}}, {{10,0,0},{14,0,0},{0,0,0}},
    {{0,1,0},{5,6,0},{0,0,0}},  {{2,3,0},{7,8,0},{0,0,0}},
    {{4,5,0},{9,10,0},{0,0,0}}, {{6,7,0},{11,12,0},{0,0,0}},
    {{8,9,0},{13,14,0},{0,0,0}},
    {{1,0,0},{7,0,0},{0,0,0}},  {{3,0,0},{9,0,0},{0,0,0}},
    {{5,0,0},{11,0,0},{0,0,0}}, {{7,0,0},{13,0,0},{0,0,0}},
    {{0,0,0},{0,0,0},{0,0,0}}
  },
  { // GEMM-B
    {{0,0,0},{4,0,0},{8,0,0}},   {{2,0,0},{6,0,0},{10,0,0}},
    {{4,0,0},{8,0,0},{12,0,0}},  {{6,0,0},{10,0,0},{14,0,0}},
    {{0,1,0},{4,5,6},{9,10,0}},  {{2,3,0},{6,7,8},{11,12,0}},
    {{4,5,0},{8,9,10},{13,14,0}},
    {{0,1,0},{5,6,7},{11,12,0}}, {{2,3,0},{7,8,9},{13,14,0}},
    {{1,0,0},{7,0,0},{13,0,0}},
    {{0,0,0},{0,0,0},{0,0,0}}, {{0,0,0},{0,0,0},{0,0,0}},
    {{0,0,0},{0,0,0},{0,0,0}}, {{0,0,0},{0,0,0},{0,0,0}},
    {{0,0,0},{0,0,0},{0,0,0}}, {{0,0,0},{0,0,0},{0,0,0}}
  }
};
__constant__ float g_wgt[2][16][3][3] = {
  { // GEMM-A: zero weight disables an entry
    {{1,0,0},{1,0,0},{0,0,0}}, {{1,0,0},{1,0,0},{0,0,0}},
    {{1,0,0},{1,0,0},{0,0,0}}, {{1,0,0},{1,0,0},{0,0,0}},
    {{1,0,0},{1,0,0},{0,0,0}}, {{1,0,0},{1,0,0},{0,0,0}},
    {{1,1,0},{1,1,0},{0,0,0}}, {{1,1,0},{1,1,0},{0,0,0}},
    {{1,1,0},{1,1,0},{0,0,0}}, {{1,1,0},{1,1,0},{0,0,0}},
    {{1,1,0},{1,1,0},{0,0,0}},
    {{1,0,0},{1,0,0},{0,0,0}}, {{1,0,0},{1,0,0},{0,0,0}},
    {{1,0,0},{1,0,0},{0,0,0}}, {{1,0,0},{1,0,0},{0,0,0}},
    {{0,0,0},{0,0,0},{0,0,0}}
  },
  { // GEMM-B: duplicated pair positions folded into weights
    {{1,0,0},{1,0,0},{1,0,0}}, {{1,0,0},{1,0,0},{1,0,0}},
    {{1,0,0},{1,0,0},{1,0,0}}, {{1,0,0},{1,0,0},{1,0,0}},
    {{2,1,0},{1,1,1},{1,2,0}}, {{2,1,0},{1,1,1},{1,2,0}},
    {{2,1,0},{1,1,1},{1,2,0}},
    {{1,2,0},{1,1,1},{2,1,0}}, {{1,2,0},{1,1,1},{2,1,0}},
    {{1,0,0},{1,0,0},{1,0,0}},
    {{0,0,0},{0,0,0},{0,0,0}}, {{0,0,0},{0,0,0},{0,0,0}},
    {{0,0,0},{0,0,0},{0,0,0}}, {{0,0,0},{0,0,0},{0,0,0}},
    {{0,0,0},{0,0,0},{0,0,0}}, {{0,0,0},{0,0,0},{0,0,0}}
  }
};
__constant__ int g_dd[2][16] = {
  {0,0,0,0,0,0, 1,1,1,1,1, 2,2,2,2, 0},
  {2,2,2,2, 3,3,3, 4,4, 5, 0,0,0,0,0,0}
};
__constant__ int g_ww[2][16] = {
  {0,1,2,3,4,5, 0,1,2,3,4, 0,1,2,3, 0},
  {0,1,2,3, 0,1,2, 0,1, 0, 0,0,0,0,0,0}
};

// -------- kernel 1: swizzle filters into WMMA B-fragment layout (f32->f16) ----
// Fragment = 32(K) x 16(N); lane l holds N=l%16, K=(l/16)*16 + j (j=0..15),
// stored so lane l's 16 halves are contiguous (1 coalesced 32B load per lane).
__global__ __launch_bounds__(256) void prep_filters_kernel(
    const float* __restrict__ f2, const float* __restrict__ f3,
    _Float16* __restrict__ w)
{
  size_t idx = (size_t)blockIdx.x * blockDim.x + threadIdx.x;
  const size_t total2 = (size_t)FRAGS2 * 512;
  const size_t total3 = (size_t)FRAGS3 * 512;
  if (idx >= total2 + total3) return;
  const int which = (idx >= total2);
  const size_t li = which ? (idx - total2) : idx;
  const int f  = (int)(li >> 9);
  const int r  = (int)(li & 511);
  const int l  = r >> 4;
  const int j  = r & 15;
  const int kl = ((l >> 4) << 4) + j;
  const int nl = l & 15;
  const int K  = (f >> 6) * 32 + kl;
  const int N  = (f & 63) * 16 + nl;
  const int t  = K >> 10;
  const int c  = K & 1023;
  const float v = which ? f3[((size_t)N * C_IN + c) * 3 + t]
                        : f2[((size_t)N * C_IN + c) * 2 + t];
  w[idx] = (_Float16)v;
}

// -------- kernel 2: zero interleaved output, pass through sum_of_squares -----
__global__ __launch_bounds__(256) void zero_and_copy_kernel(
    float* __restrict__ out, const float* __restrict__ sos)
{
  const size_t n4 = (size_t)BATCH * O_OUT * OUTW / 4;
  size_t i = (size_t)blockIdx.x * blockDim.x + threadIdx.x;
  const size_t stride = (size_t)gridDim.x * blockDim.x;
  const float4 z = make_float4(0.f, 0.f, 0.f, 0.f);
  for (size_t k = i; k < n4; k += stride) ((float4*)out)[k] = z;
  if (i < BATCH) out[(size_t)BATCH * O_OUT * OUTW + i] = sos[i];
}

// -------- kernel 3/4: fused gather + f16 WMMA GEMM with LDS-staged A -------
// Block: 256 threads = 8 waves as 2(M) x 4(N). Block tile: M=32, N=512.
// Per 32-wide K-chunk the block cooperatively gathers the 32x32 A tile from x
// (branch-free weighted sum, f32->f16) straight into WMMA-fragment order in
// LDS (double buffered, one barrier per chunk). Each wave then reads its A
// fragment with one 32B ds load per lane and runs 8 v_wmma_f32_16x16x32_f16
// against 8 pre-swizzled, L2-resident B fragments.
template <int WHICH, int SLOTS, int TCOUNT, int NP>
__global__ __launch_bounds__(256) void gemm_wmma_kernel(
    const float* __restrict__ x, const _Float16* __restrict__ wfrag,
    float* __restrict__ out, float invnorm)
{
  __shared__ _Float16 sA[2][1024];   // [buffer][32 rows x 32 K], fragment order

  const int lane = threadIdx.x & 31;
  const int wave = threadIdx.x >> 5;
  const int msub = wave & 1;
  const int nsub = wave >> 1;
  const int ml   = lane & 15;

  // ---- staging role: thread covers (row_st, kl = kb..kb+3) of the A tile ----
  const int row_st = threadIdx.x >> 3;          // 0..31
  const int kb     = (threadIdx.x & 7) * 4;     // 0,4,...,28
  const int rowGst = blockIdx.x * 32 + row_st;
  const int b_st   = rowGst / SLOTS;
  const int s_st   = rowGst - b_st * SLOTS;
  const float* __restrict__ xst = x + (size_t)b_st * (C_IN * 18);
  // fragment-order LDS index: lane = h*16 + (row%16), halves j contiguous
  const int h_st = kb >> 4;
  const int jb   = kb & 15;
  const int lidx = (((row_st >> 4) * 32) + h_st * 16 + (row_st & 15)) * 16 + jb;

  const int nfbase = blockIdx.y * 32 + nsub * 8;   // 8 N-fragments per wave

  v8f acc[8] = {v8f{}, v8f{}, v8f{}, v8f{}, v8f{}, v8f{}, v8f{}, v8f{}};

  for (int t = 0; t < TCOUNT; ++t) {
    const int   p0 = g_pos[WHICH][s_st][t][0];
    const int   p1 = g_pos[WHICH][s_st][t][1];
    const int   p2 = g_pos[WHICH][s_st][t][2];
    const float w0 = g_wgt[WHICH][s_st][t][0];
    const float w1 = g_wgt[WHICH][s_st][t][1];
    const float w2 = g_wgt[WHICH][s_st][t][2];

    for (int kfl = 0; kfl < 32; ++kfl) {
      const int kf  = t * 32 + kfl;
      const int buf = kf & 1;

      // ---- cooperative branch-free gather of 4 A elements -> LDS ----
      {
        const int cbase = kfl * 32;
        v4h hv;
#pragma unroll
        for (int q = 0; q < 4; ++q) {
          const float* __restrict__ px = xst + (size_t)(cbase + kb + q) * 18;
          float s = w0 * px[p0] + w1 * px[p1];
          if (NP == 3) s += w2 * px[p2];
          hv[q] = (_Float16)s;
        }
        *(v4h*)&sA[buf][lidx] = hv;   // 8B ds_store, fragment order
      }
      __syncthreads();

      // ---- one 32B ds read per lane gives the whole A fragment ----
      const v16h a = *(const v16h*)&sA[buf][(msub * 32 + lane) * 16];

      // ---- 8 pre-swizzled B fragments (32B/lane, coalesced, L2-hot) ----
      const _Float16* __restrict__ wb =
          wfrag + ((size_t)(kf * 64 + nfbase)) * 512 + lane * 16;
      __builtin_prefetch(wb + 64 * 512, 0, 1);   // next K-chunk
#pragma unroll
      for (int i = 0; i < 8; ++i) {
        const v16h bf = *(const v16h*)(wb + i * 512);
        acc[i] = __builtin_amdgcn_wmma_f32_16x16x32_f16(
            false, a, false, bf, (short)0, acc[i], false, false);
      }
      // next iteration stages the other buffer; the following barrier
      // protects re-use of this one.
    }
  }

  // Epilogue: C/D layout -> lane half h, VGPR r holds M = r + 8*h, N = lane%16.
  const int h = lane >> 4;
#pragma unroll
  for (int i = 0; i < 8; ++i) {
    const int o = (nfbase + i) * 16 + ml;
#pragma unroll
    for (int r = 0; r < 8; ++r) {
      const int rowC = blockIdx.x * 32 + msub * 16 + 8 * h + r;
      const int bb = rowC / SLOTS;
      const int ss = rowC - bb * SLOTS;
      const int col = g_ww[WHICH][ss] * 6 + g_dd[WHICH][ss];  // w*6 + (len-4)
      const size_t oi = ((size_t)bb * O_OUT + o) * OUTW + col;
      out[oi] += acc[i][r] * invnorm;   // distinct slot sets per kernel; zeroed base
    }
  }
}

extern "C" void kernel_launch(void* const* d_in, const int* in_sizes, int n_in,
                              void* d_out, int out_size, void* d_ws, size_t ws_size,
                              hipStream_t stream)
{
  (void)in_sizes; (void)n_in; (void)out_size; (void)ws_size;
  const float* x   = (const float*)d_in[0];
  const float* sos = (const float*)d_in[1];
  const float* f2  = (const float*)d_in[2];
  const float* f3  = (const float*)d_in[3];
  float* out = (float*)d_out;
  _Float16* w = (_Float16*)d_ws;     // needs (FRAGS2+FRAGS3)*512*2 = 10.5 MB

  const size_t totalw = (size_t)(FRAGS2 + FRAGS3) * 512;
  prep_filters_kernel<<<(unsigned)((totalw + 255) / 256), 256, 0, stream>>>(f2, f3, w);
  zero_and_copy_kernel<<<8192, 256, 0, stream>>>(out, sos);

  const float inv2 = 0.022097086912079612f;  // 1/sqrt(2*1024)
  const float inv3 = 0.018042195912175804f;  // 1/sqrt(3*1024)
  // GEMM-A: M = 2048*15 = 30720 -> 960 M-blocks; N = 1024 -> 2 N-blocks
  gemm_wmma_kernel<0, 15, 2, 2><<<dim3(960, 2), 256, 0, stream>>>(x, w, out, inv2);
  // GEMM-B: M = 2048*10 = 20480 -> 640 M-blocks
  gemm_wmma_kernel<1, 10, 3, 3><<<dim3(640, 2), 256, 0, stream>>>(
      x, w + (size_t)FRAGS2 * 512, out, inv3);
}
